// GraphConvolution_79800492359867
// MI455X (gfx1250) — matrix operand order repaired and checked by reading, not compiled, not run on hardware
//
#include <hip/hip_runtime.h>
#include <hip/hip_bf16.h>
#include <hip/hip_fp16.h>

// ---------------------------------------------------------------------------
// GraphConvolution for MI455X (gfx1250, wave32, WMMA)
//   1) dropout(x) -> f16 copy xh            [N, 256]
//   2) W (f32, 256x128) -> Wt f16 (128x256) (transposed for contiguous B frags)
//   3) pre_sup = xh @ W via v_wmma_f32_16x16x32_f16, f32 accum  [N, 128]
//   4) out = 0; for each edge: out[row] += val * pre_sup[col]   (HW f32 atomics)
//   5) out = relu(out)
// ---------------------------------------------------------------------------

typedef __attribute__((ext_vector_type(16))) _Float16 v16h;
typedef __attribute__((ext_vector_type(8)))  _Float16 v8h;
typedef __attribute__((ext_vector_type(4)))  _Float16 v4h;
typedef __attribute__((ext_vector_type(8)))  float    v8f;

#define D_DIM 256
#define O_DIM 128
#define KEEP  0.9f

// Stateless per-element hash -> uniform [0,1). (JAX threefry not reproducible
// here; deterministic per index so graph replays validate consistently.)
__device__ __forceinline__ float hash_u01(unsigned int i) {
  unsigned int x = i * 0x9E3779B9u ^ 0xA5A5A5A5u;
  x ^= x >> 16; x *= 0x7feb352du;
  x ^= x >> 15; x *= 0x846ca68bu;
  x ^= x >> 16;
  return (float)(x >> 8) * (1.0f / 16777216.0f);
}

// ---- 1) dropout + f32 -> f16 (vectorized: float4 in, 4x f16 out) ----------
__global__ void dropout_cast_kernel(const float* __restrict__ x,
                                    _Float16* __restrict__ xh, int n4) {
  int i = blockIdx.x * blockDim.x + threadIdx.x;
  if (i >= n4) return;
  float4 v = ((const float4*)x)[i];
  const float inv = 1.0f / KEEP;
  unsigned int base = (unsigned int)i * 4u;
  v4h o;
  o[0] = (_Float16)((hash_u01(base + 0u) < KEEP) ? v.x * inv : 0.0f);
  o[1] = (_Float16)((hash_u01(base + 1u) < KEEP) ? v.y * inv : 0.0f);
  o[2] = (_Float16)((hash_u01(base + 2u) < KEEP) ? v.z * inv : 0.0f);
  o[3] = (_Float16)((hash_u01(base + 3u) < KEEP) ? v.w * inv : 0.0f);
  ((v4h*)xh)[i] = o;
}

// ---- 2) W[k][o] (f32) -> Wt[o][k] (f16) ------------------------------------
__global__ void wcast_kernel(const float* __restrict__ w,
                             _Float16* __restrict__ wt) {
  int i = blockIdx.x * blockDim.x + threadIdx.x;   // over D*O
  if (i >= D_DIM * O_DIM) return;
  int k = i / O_DIM;
  int o = i % O_DIM;
  wt[o * D_DIM + k] = (_Float16)w[i];
}

// ---- 3) WMMA GEMM: one 16x16 output tile per wave; 8 waves = 16x128 stripe -
// A fragment layout (16-bit, 16x32):  lanes 0-15: M=lane, elems = K{0..7,16..23}
//                                     lanes16-31: M=lane-16, K{8..15,24..31}
// B fragment layout (16-bit, 32x16):  lanes 0-15: N=lane, elems = K{0..7,16..23}
//                                     lanes16-31: N=lane-16, K{8..15,24..31}
// C layout (f32 16x16): VGPR v: lanes0-15 -> (M=v,N=lane); lanes16-31 -> (M=8+v)
__global__ void __launch_bounds__(256)
gemm_wmma_kernel(const _Float16* __restrict__ xh,
                 const _Float16* __restrict__ wt,
                 float* __restrict__ pre, int N) {
  const int wave = threadIdx.x >> 5;     // 0..7 -> n-tile
  const int lane = threadIdx.x & 31;
  const int half = lane >> 4;            // lane group 0/1
  const int l    = lane & 15;
  const int m0 = blockIdx.x * 16;
  const int n0 = wave * 16;
  if (m0 >= N) return;

  const int arow = (m0 + l) < N ? (m0 + l) : (N - 1);
  const _Float16* aptr = xh + (size_t)arow * D_DIM + half * 8;
  const _Float16* bptr = wt + (size_t)(n0 + l) * D_DIM + half * 8;

  v8f acc = {};
#pragma unroll
  for (int kc = 0; kc < D_DIM; kc += 32) {
    v8h alo = *(const v8h*)(aptr + kc);        // K = kc + half*8 + [0..7]
    v8h ahi = *(const v8h*)(aptr + kc + 16);   // K = kc + 16 + half*8 + [0..7]
    v16h a, b;
    v8h blo = *(const v8h*)(bptr + kc);        // K = kc + half*8 + [0..7]
    v8h bhi = *(const v8h*)(bptr + kc + 16);   // K = kc + 16 + half*8 + [0..7]
#pragma unroll
    for (int t = 0; t < 8; ++t) { a[t] = alo[t]; a[8 + t] = ahi[t]; b[t] = blo[t]; b[8 + t] = bhi[t]; }
    acc = __builtin_amdgcn_wmma_f32_16x16x32_f16(
        /*neg_a=*/false, a, /*neg_b=*/false, b,
        /*c_mod=*/(short)0, acc, /*reuse_a=*/false, /*reuse_b=*/false);
  }

  float* orow = pre + (size_t)(m0 + half * 8) * O_DIM + n0 + l;
#pragma unroll
  for (int v = 0; v < 8; ++v) {
    int m = m0 + half * 8 + v;
    if (m < N) orow[(size_t)v * O_DIM] = acc[v];
  }
}

// ---- 4a) zero output (harness poisons d_out; atomics need clean zeros) -----
__global__ void zero_kernel(float* __restrict__ out, int n) {
  int i = blockIdx.x * blockDim.x + threadIdx.x;
  if (i < n) out[i] = 0.0f;
}

// ---- 4b) edge scatter: one wave32 per edge, float4 per lane (128 cols) -----
__global__ void __launch_bounds__(256)
spmm_edge_kernel(const float* __restrict__ pre,
                 const int* __restrict__ erow, const int* __restrict__ ecol,
                 const float* __restrict__ eval, float* __restrict__ out, int E) {
  int wid  = (blockIdx.x * blockDim.x + threadIdx.x) >> 5;
  int lane = threadIdx.x & 31;
  if (wid >= E) return;
  const int   r = erow[wid];
  const int   c = ecol[wid];
  const float v = eval[wid];
  float4 m = ((const float4*)(pre + (size_t)c * O_DIM))[lane];
  float* dst = out + (size_t)r * O_DIM + lane * 4;
  unsafeAtomicAdd(dst + 0, v * m.x);
  unsafeAtomicAdd(dst + 1, v * m.y);
  unsafeAtomicAdd(dst + 2, v * m.z);
  unsafeAtomicAdd(dst + 3, v * m.w);
}

// ---- 5) ReLU ---------------------------------------------------------------
__global__ void relu_kernel(float* __restrict__ out, int n) {
  int i = blockIdx.x * blockDim.x + threadIdx.x;
  if (i < n) out[i] = fmaxf(out[i], 0.0f);
}

extern "C" void kernel_launch(void* const* d_in, const int* in_sizes, int n_in,
                              void* d_out, int out_size, void* d_ws, size_t ws_size,
                              hipStream_t stream) {
  const float* x        = (const float*)d_in[0];
  const int*   edge_row = (const int*)d_in[1];
  const int*   edge_col = (const int*)d_in[2];
  const float* edge_val = (const float*)d_in[3];
  const float* weight   = (const float*)d_in[4];

  const int N = in_sizes[0] / D_DIM;   // 100000
  const int E = in_sizes[1];           // 1600000

  // Workspace carving: xh (N*256 f16) | Wt (128*256 f16) | pre_sup (N*128 f32)
  char* ws = (char*)d_ws;
  _Float16* xh = (_Float16*)ws;
  size_t off = (size_t)N * D_DIM * sizeof(_Float16);
  _Float16* wt = (_Float16*)(ws + off);
  off += (size_t)D_DIM * O_DIM * sizeof(_Float16);
  off = (off + 255) & ~(size_t)255;
  float* pre = (float*)(ws + off);

  const int n4 = (N * D_DIM) / 4;
  dropout_cast_kernel<<<(n4 + 255) / 256, 256, 0, stream>>>(x, xh, n4);
  wcast_kernel<<<(D_DIM * O_DIM + 255) / 256, 256, 0, stream>>>(weight, wt);

  const int row_tiles = (N + 15) / 16;   // 6250 blocks, 8 waves each
  gemm_wmma_kernel<<<row_tiles, 256, 0, stream>>>(xh, wt, pre, N);

  zero_kernel<<<(out_size + 255) / 256, 256, 0, stream>>>((float*)d_out, out_size);

  const int edge_blocks = (E + 7) / 8;   // 8 edges (waves) per 256-thread block
  spmm_edge_kernel<<<edge_blocks, 256, 0, stream>>>(pre, edge_row, edge_col,
                                                    edge_val, (float*)d_out, E);

  relu_kernel<<<(out_size + 255) / 256, 256, 0, stream>>>((float*)d_out, out_size);
}
